// GCN_16896401342481
// MI455X (gfx1250) — matrix operand order repaired and checked by reading, not compiled, not run on hardware
//
#include <hip/hip_runtime.h>
#include <cmath>

typedef __attribute__((ext_vector_type(16))) _Float16 v16h;
typedef __attribute__((ext_vector_type(8)))  _Float16 v8h;
typedef __attribute__((ext_vector_type(8)))  float    v8f;

#define N_NODES 50000
#define N_EDGES 800000
#define F_IN    128
#define F_HID   256
#define F_OUT   40
#define F_OUT_P 48   // padded to 3 WMMA N-tiles
#define NB1     4    // N-tiles per wave in GEMM1 (16x64 strip)

// ---------------- utility kernels ----------------

__global__ void k_zero(float* __restrict__ p, int n) {
    int i = blockIdx.x * blockDim.x + threadIdx.x;
    if (i < n) p[i] = 0.0f;
}

__global__ void k_deg(const int* __restrict__ src, const int* __restrict__ dst,
                      float* __restrict__ deg_out, float* __restrict__ deg_in) {
    int e = blockIdx.x * blockDim.x + threadIdx.x;
    if (e < N_EDGES) {
        atomicAdd(&deg_out[src[e]], 1.0f);
        atomicAdd(&deg_in[dst[e]], 1.0f);
    }
}

__global__ void k_norm(const float* __restrict__ deg, float* __restrict__ nrm, int n) {
    int i = blockIdx.x * blockDim.x + threadIdx.x;
    if (i < n) nrm[i] = rsqrtf(fmaxf(deg[i], 1.0f));
}

__global__ void k_cvt_w1(const float* __restrict__ W1, _Float16* __restrict__ W1h) {
    int i = blockIdx.x * blockDim.x + threadIdx.x;
    if (i < F_IN * F_HID) W1h[i] = (_Float16)W1[i];
}

// W2 (256x40) -> padded f16 (256x48), zero-filled pad columns
__global__ void k_cvt_w2(const float* __restrict__ W2, _Float16* __restrict__ W2h) {
    int i = blockIdx.x * blockDim.x + threadIdx.x;
    if (i < F_HID * F_OUT_P) {
        int r = i / F_OUT_P, c = i - r * F_OUT_P;
        W2h[i] = (c < F_OUT) ? (_Float16)W2[r * F_OUT + c] : (_Float16)0.0f;
    }
}

// ---------------- SpMM layer 1 (128 feats/edge) ----------------
// agg1[dst] += features[src] * norm_src[src]
__global__ void k_spmm1(const float* __restrict__ feat, const float* __restrict__ norm_src,
                        const int* __restrict__ src, const int* __restrict__ dst,
                        float* __restrict__ agg1) {
    int idx = blockIdx.x * blockDim.x + threadIdx.x;  // < 800000*128 = 102.4M
    int e = idx >> 7;
    int f = idx & 127;
    if (e < N_EDGES) {
        int s = src[e];
        float v = feat[(size_t)s * F_IN + f] * norm_src[s];
        atomicAdd(&agg1[(size_t)dst[e] * F_IN + f], v);
    }
}

// ---------------- GEMM1: relu((norm_dst . agg1) @ W1 + b1) -> h (f16) ------
// one wave per 16x64 output strip: A fragment built once per K-step,
// reused across NB1 WMMAs (amortizes agg1 reads and f32->f16 conversion)
__global__ void k_gemm1(const float* __restrict__ agg1, const float* __restrict__ norm_dst,
                        const _Float16* __restrict__ W1h, const float* __restrict__ b1,
                        _Float16* __restrict__ h_h) {
    int wave = (blockIdx.x * blockDim.x + threadIdx.x) >> 5;
    int lane = threadIdx.x & 31;
    const int NW = (F_HID / 16) / NB1;   // 4 wave-columns
    int mt = wave / NW;
    int nw = wave - mt * NW;
    if (mt >= N_NODES / 16) return;      // wave-uniform: EXEC all-1s for WMMA
    int m0 = mt * 16, n0 = nw * (16 * NB1);
    int row = lane & 15, half = lane >> 4;

    float nd = norm_dst[m0 + row];
    const float* arow = agg1 + (size_t)(m0 + row) * F_IN;
    v8f acc[NB1] = {};
    #pragma unroll
    for (int k0 = 0; k0 < F_IN; k0 += 32) {
        // A fragment (16-bit A 16x32 layout): lane row = lane&15,
        // e0..7 -> K = k0 + 8*half + e ; e8..15 -> K = k0 + 16 + 8*half + (e-8)
        v16h a;
        const float* p = arow + k0 + 8 * half;
        #pragma unroll
        for (int e = 0; e < 8; ++e) a[e]     = (_Float16)(p[e]      * nd);
        #pragma unroll
        for (int e = 0; e < 8; ++e) a[8 + e] = (_Float16)(p[16 + e] * nd);
        // B fragments: lane col N = lane&15, elements K in same order as A
        #pragma unroll
        for (int j = 0; j < NB1; ++j) {
            v16h b;
            #pragma unroll
            for (int e = 0; e < 8; ++e) {
                b[e]     = W1h[(size_t)(k0 + 8 * half + e) * F_HID + n0 + 16 * j + row];
                b[8 + e] = W1h[(size_t)(k0 + 16 + 8 * half + e) * F_HID + n0 + 16 * j + row];
            }
            acc[j] = __builtin_amdgcn_wmma_f32_16x16x32_f16(false, a, false, b,
                                                            (short)0, acc[j], false, false);
        }
    }
    // D layout: N = lane&15, M = r + 8*half
    #pragma unroll
    for (int j = 0; j < NB1; ++j) {
        int col = n0 + 16 * j + row;
        float bias = b1[col];
        #pragma unroll
        for (int r = 0; r < 8; ++r) {
            int m = m0 + r + 8 * half;
            float v = acc[j][r] + bias;
            v = v > 0.0f ? v : 0.0f;
            h_h[(size_t)m * F_HID + col] = (_Float16)v;
        }
    }
}

// ---------------- GEMM2: ts = norm_src . (h @ W2)  (40 cols) --------------
// one wave per 16x48 strip (full output width): A read once per K-step
__global__ void k_gemm2(const _Float16* __restrict__ h_h, const float* __restrict__ norm_src,
                        const _Float16* __restrict__ W2h, float* __restrict__ ts) {
    int wave = (blockIdx.x * blockDim.x + threadIdx.x) >> 5;
    int lane = threadIdx.x & 31;
    if (wave >= N_NODES / 16) return;    // wave-uniform
    int m0 = wave * 16;
    int row = lane & 15, half = lane >> 4;

    const _Float16* arow = h_h + (size_t)(m0 + row) * F_HID;
    v8f acc[3] = {};
    #pragma unroll
    for (int k0 = 0; k0 < F_HID; k0 += 32) {
        v16h a;
        const _Float16* p = arow + k0 + 8 * half;
        v8h lo = *(const v8h*)(p);
        v8h hi = *(const v8h*)(p + 16);
        #pragma unroll
        for (int e = 0; e < 8; ++e) { a[e] = lo[e]; a[8 + e] = hi[e]; }
        #pragma unroll
        for (int j = 0; j < 3; ++j) {
            v16h b;
            #pragma unroll
            for (int e = 0; e < 8; ++e) {
                b[e]     = W2h[(size_t)(k0 + 8 * half + e) * F_OUT_P + 16 * j + row];
                b[8 + e] = W2h[(size_t)(k0 + 16 + 8 * half + e) * F_OUT_P + 16 * j + row];
            }
            acc[j] = __builtin_amdgcn_wmma_f32_16x16x32_f16(false, a, false, b,
                                                            (short)0, acc[j], false, false);
        }
    }
    #pragma unroll
    for (int j = 0; j < 3; ++j) {
        int col = 16 * j + row;
        if (col < F_OUT) {
            #pragma unroll
            for (int r = 0; r < 8; ++r) {
                int m = m0 + r + 8 * half;
                ts[(size_t)m * F_OUT + col] = acc[j][r] * norm_src[m];
            }
        }
    }
}

// ---------------- SpMM layer 2 (40 feats/edge) ----------------
__global__ void k_spmm2(const float* __restrict__ ts, const int* __restrict__ src,
                        const int* __restrict__ dst, float* __restrict__ agg2) {
    int idx = blockIdx.x * blockDim.x + threadIdx.x;  // < 800000*40 = 32M
    int e = idx / F_OUT;
    int f = idx - e * F_OUT;
    if (e < N_EDGES) {
        atomicAdd(&agg2[(size_t)dst[e] * F_OUT + f], ts[(size_t)src[e] * F_OUT + f]);
    }
}

// ---------------- final: z = norm_dst.agg2 + b2 ; log_softmax -------------
// one wave32 per node; lanes 0..31 cover class c=lane, lanes 0..7 also c=lane+32
__global__ void k_lsm(const float* __restrict__ agg2, const float* __restrict__ norm_dst,
                      const float* __restrict__ b2, float* __restrict__ out) {
    int wid = (blockIdx.x * blockDim.x + threadIdx.x) >> 5;
    int lane = threadIdx.x & 31;
    if (wid >= N_NODES) return;
    float nd = norm_dst[wid];
    const float* row = agg2 + (size_t)wid * F_OUT;
    float za = row[lane] * nd + b2[lane];
    float zb = (lane < F_OUT - 32) ? (row[32 + lane] * nd + b2[32 + lane]) : -3.4e38f;
    float mx = fmaxf(za, zb);
    #pragma unroll
    for (int off = 16; off; off >>= 1) mx = fmaxf(mx, __shfl_xor(mx, off, 32));
    float s = expf(za - mx) + ((lane < F_OUT - 32) ? expf(zb - mx) : 0.0f);
    #pragma unroll
    for (int off = 16; off; off >>= 1) s += __shfl_xor(s, off, 32);
    float lse = mx + logf(s);
    float* orow = out + (size_t)wid * F_OUT;
    orow[lane] = za - lse;
    if (lane < F_OUT - 32) orow[32 + lane] = zb - lse;
}

// ---------------- launch ----------------

static inline size_t al256(size_t x) { return (x + 255) & ~(size_t)255; }

extern "C" void kernel_launch(void* const* d_in, const int* in_sizes, int n_in,
                              void* d_out, int out_size, void* d_ws, size_t ws_size,
                              hipStream_t stream) {
    const float* features = (const float*)d_in[0];   // 50000 x 128
    const float* W1       = (const float*)d_in[1];   // 128 x 256
    const float* b1       = (const float*)d_in[2];   // 256
    const float* W2       = (const float*)d_in[3];   // 256 x 40
    const float* b2       = (const float*)d_in[4];   // 40
    const int*   src      = (const int*)d_in[5];     // 800000
    const int*   dst      = (const int*)d_in[6];     // 800000
    float* out = (float*)d_out;                      // 50000 x 40

    // workspace layout
    char* ws = (char*)d_ws;
    size_t off = 0;
    float*     deg_out  = (float*)(ws + off);  off += al256(N_NODES * 4);
    float*     deg_in   = (float*)(ws + off);  off += al256(N_NODES * 4);
    float*     norm_src = (float*)(ws + off);  off += al256(N_NODES * 4);
    float*     norm_dst = (float*)(ws + off);  off += al256(N_NODES * 4);
    _Float16*  W1h      = (_Float16*)(ws + off); off += al256(F_IN * F_HID * 2);
    _Float16*  W2h      = (_Float16*)(ws + off); off += al256(F_HID * F_OUT_P * 2);
    float*     agg1     = (float*)(ws + off);  off += al256((size_t)N_NODES * F_IN * 4);   // 25.6 MB
    _Float16*  h_h      = (_Float16*)(ws + off); off += al256((size_t)N_NODES * F_HID * 2); // 25.6 MB
    // layer-2 scratch aliases agg1 region (agg1 is dead after k_gemm1)
    float* ts   = agg1;                                            // 8 MB
    float* agg2 = (float*)((char*)agg1 + al256((size_t)N_NODES * F_OUT * 4) + 4 * 1024 * 1024); // clear of ts
    (void)in_sizes; (void)n_in; (void)out_size; (void)ws_size;

    const int B = 256;

    // degrees + norms
    k_zero<<<(N_NODES + B - 1) / B, B, 0, stream>>>(deg_out, N_NODES);
    k_zero<<<(N_NODES + B - 1) / B, B, 0, stream>>>(deg_in,  N_NODES);
    k_deg<<<(N_EDGES + B - 1) / B, B, 0, stream>>>(src, dst, deg_out, deg_in);
    k_norm<<<(N_NODES + B - 1) / B, B, 0, stream>>>(deg_out, norm_src, N_NODES);
    k_norm<<<(N_NODES + B - 1) / B, B, 0, stream>>>(deg_in,  norm_dst, N_NODES);

    // weight conversion
    k_cvt_w1<<<(F_IN * F_HID + B - 1) / B, B, 0, stream>>>(W1, W1h);
    k_cvt_w2<<<(F_HID * F_OUT_P + B - 1) / B, B, 0, stream>>>(W2, W2h);

    // layer 1
    k_zero<<<((N_NODES * F_IN) + B - 1) / B, B, 0, stream>>>(agg1, N_NODES * F_IN);
    {
        long long tot = (long long)N_EDGES * F_IN;          // 102.4M threads
        k_spmm1<<<(int)((tot + B - 1) / B), B, 0, stream>>>(features, norm_src, src, dst, agg1);
    }
    {
        int waves = (N_NODES / 16) * ((F_HID / 16) / NB1);  // 12500 wave-strips
        k_gemm1<<<(waves * 32 + B - 1) / B, B, 0, stream>>>(agg1, norm_dst, W1h, b1, h_h);
    }

    // layer 2 (GEMM before SpMM: 40-wide scatter instead of 256-wide)
    {
        int waves = N_NODES / 16;                           // 3125 wave-strips
        k_gemm2<<<(waves * 32 + B - 1) / B, B, 0, stream>>>(h_h, norm_src, W2h, ts);
    }
    k_zero<<<((N_NODES * F_OUT) + B - 1) / B, B, 0, stream>>>(agg2, N_NODES * F_OUT);
    {
        long long tot = (long long)N_EDGES * F_OUT;         // 32M threads
        k_spmm2<<<(int)((tot + B - 1) / B), B, 0, stream>>>(ts, src, dst, agg2);
    }

    // bias + log_softmax
    k_lsm<<<((N_NODES * 32) + B - 1) / B, B, 0, stream>>>(agg2, norm_dst, b2, out);
}
